// Obiwan_84335977825044
// MI455X (gfx1250) — matrix operand order repaired and checked by reading, not compiled, not run on hardware
//
#include <hip/hip_runtime.h>
#include <hip/hip_bf16.h>
#include <math.h>

// ---------------------------------------------------------------------------
// Problem constants (B=32 molecules, N=32 atoms, T=465 triplets per atom)
// ---------------------------------------------------------------------------
#define NATOM      32
#define NBATCH     32
#define TRI        465                    // (N-1)(N-2)/2
#define ROWS_TOTAL (NBATCH * NATOM * TRI) // 476160
#define TM         128                    // triplet rows per workgroup
#define NWAVE      8                      // 256 threads / wave32
#define OUTF       256                    // final feature width
#define CUTOFF     3.5f
#define PI_F       3.14159265358979323846f

typedef __attribute__((ext_vector_type(16))) _Float16 v16h;
typedef __attribute__((ext_vector_type(8)))  _Float16 v8h;
typedef __attribute__((ext_vector_type(8)))  float    v8f;

// ---------------------------------------------------------------------------
// LDS layout (bytes). Weights stored transposed f16: wT[n*Kpad + k].
// ---------------------------------------------------------------------------
#define OW0 0
#define OW1 2048
#define OW2 6144
#define OW3 10240
#define OW4 14336
#define OW5 18432
#define OW6 26624
#define W_HALVES 59392
#define BIAS_BYTE   (W_HALVES * 2)        // 118784 : 704 floats
#define SMOOTH_BYTE (BIAS_BYTE + 704*4)   // 121600 : 128 floats
#define ROWOUT_BYTE (SMOOTH_BYTE + 128*4) // 122112 : 128 ints
#define GI_BYTE     (ROWOUT_BYTE + 128*4) // 122624 : 128 ints (group flag)
#define ACC_BYTE    (GI_BYTE + 128*4)     // 123136 : 512 floats (2 groups x 256)
#define ACT_BYTE    (ACC_BYTE + 512*4)    // 125184
// per-wave scratch: in(16x32)=1024B, X=2048B, H=2048B, BLK=2048B, B5(16x128)=4096B
#define ACT_PER_WAVE 11264
#define SMEM_TOTAL  (ACT_BYTE + NWAVE * ACT_PER_WAVE) // 215296 bytes

__device__ __forceinline__ void lds_fence() {
    asm volatile("s_wait_dscnt 0" ::: "memory");
}

// Branch-free tanh: one v_exp_f32 + one v_rcp_f32, no EXEC-mask branches.
__device__ __forceinline__ float fast_tanh(float x) {
    const float cx = fminf(fmaxf(x, -9.0f), 9.0f);
    const float e  = __expf(2.0f * cx);
    return (e - 1.0f) * __builtin_amdgcn_rcpf(e + 1.0f);
}

// Accumulate one 16x16 output block over the K dimension with f16 WMMA.
// A-frag: lane holds K = {8g..8g+7, 16+8g..23+8g}  -> two b128 LDS loads
// B-frag (wT transposed): lane holds K = {8g..8g+7, 16+8g..23+8g} -> two b128 LDS loads
__device__ __forceinline__ v8f wmma_accum(const _Float16* ain, int Kdim,
                                          const _Float16* wcol,
                                          int g, int mlo, v8f acc) {
#pragma unroll
    for (int kb = 0; kb < Kdim; kb += 32) {
        const _Float16* ap = ain + mlo * Kdim + kb + 8 * g;
        v8h alo = *(const v8h*)(ap);
        v8h ahi = *(const v8h*)(ap + 16);
        const _Float16* bp = wcol + kb + 8 * g;
        v8h blo = *(const v8h*)(bp);
        v8h bhi = *(const v8h*)(bp + 16);
        v16h A, Bm;
#pragma unroll
        for (int q = 0; q < 8; ++q) {
            A[q] = alo[q];  A[q + 8] = ahi[q];
            Bm[q] = blo[q]; Bm[q + 8] = bhi[q];
        }
        acc = __builtin_amdgcn_wmma_f32_16x16x32_f16(false, A, false, Bm,
                                                     (short)0, acc, false, false);
    }
    return acc;
}

// One dense layer on a 16-row tile: aout = tanh(ain @ W + b) (+ resid).
__device__ __forceinline__ void wmma_layer(const _Float16* ain, int Kdim,
                                           _Float16* aout, int Ndim,
                                           const _Float16* wT, const float* bias,
                                           const _Float16* resid, int lane) {
    const int g = lane >> 4, mlo = lane & 15;
    for (int cb = 0; cb < Ndim; cb += 16) {
        const int n = cb + mlo;
        v8f acc = {};
        acc = wmma_accum(ain, Kdim, wT + n * Kdim, g, mlo, acc);
        const float bn = bias[n];
#pragma unroll
        for (int r = 0; r < 8; ++r) {
            const int M = r + 8 * g;
            float v = fast_tanh(acc[r] + bn);
            if (resid) v += (float)resid[M * Ndim + n];
            aout[M * Ndim + n] = (_Float16)v;
        }
    }
    lds_fence();
}

// Final layer: tanh, scale by smooth weight, reduce the 16 rows into the
// per-WG 2-group LDS accumulator (a 128-row tile spans <= 2 (b,i) groups).
__device__ __forceinline__ void wmma_final(const _Float16* ain, const _Float16* wT,
                                           const float* bias, const float* smoothw,
                                           const int* giw, float* accbuf, int lane) {
    const int g = lane >> 4, mlo = lane & 15;
    for (int cb = 0; cb < OUTF; cb += 16) {
        const int n = cb + mlo;
        v8f acc = {};
        acc = wmma_accum(ain, 128, wT + n * 128, g, mlo, acc);
        const float bn = bias[n];
        float s0 = 0.0f, s1 = 0.0f;
#pragma unroll
        for (int r = 0; r < 8; ++r) {
            const int M = r + 8 * g;
            const float v = fast_tanh(acc[r] + bn) * smoothw[M];
            if (giw[M]) s1 += v; else s0 += v;
        }
        atomicAdd(accbuf + n, s0);          // ds_add_f32
        atomicAdd(accbuf + OUTF + n, s1);   // ds_add_f32
    }
}

__device__ __forceinline__ float f_cut(float d) {
    return 0.5f * __cosf(PI_F * d / CUTOFF) + 0.5f;
}

__global__ void __launch_bounds__(256, 1)
aev_triplet_kernel(const float* __restrict__ D, const float* __restrict__ Z,
                   const float* W0, const float* b0, const float* W1, const float* b1,
                   const float* W2, const float* b2, const float* W3, const float* b3,
                   const float* W4, const float* b4, const float* W5, const float* b5,
                   const float* W6, const float* b6, float* __restrict__ out) {
    extern __shared__ __align__(16) char smem[];
    _Float16* wT        = (_Float16*)smem;
    float*    biasL     = (float*)(smem + BIAS_BYTE);
    float*    smoothArr = (float*)(smem + SMOOTH_BYTE);
    int*      rowoutArr = (int*)(smem + ROWOUT_BYTE);
    int*      giArr     = (int*)(smem + GI_BYTE);
    float*    accOut    = (float*)(smem + ACC_BYTE);
    char*     actBase   = smem + ACT_BYTE;

    const int tid = threadIdx.x;

    // ---- Stage 0: weights -> LDS, f16, transposed, K zero-padded --------
    {
        const int   LK[7]  = {9, 64, 64, 64, 64, 64, 128};
        const int   LN[7]  = {64, 64, 64, 64, 64, 128, 256};
        const int   LKP[7] = {32, 64, 64, 64, 64, 64, 128};
        const int   LWO[7] = {OW0, OW1, OW2, OW3, OW4, OW5, OW6};
        const int   LBO[7] = {0, 64, 128, 192, 256, 320, 448};
        const float* LW[7] = {W0, W1, W2, W3, W4, W5, W6};
        const float* LB[7] = {b0, b1, b2, b3, b4, b5, b6};
#pragma unroll
        for (int l = 0; l < 7; ++l) {
            const int K = LK[l], Nn = LN[l], Kp = LKP[l], total = Nn * Kp;
            _Float16* dst = wT + LWO[l];
            const float* Wg = LW[l];
            for (int idx = tid; idx < total; idx += 256) {
                const int n = idx / Kp, k = idx - n * Kp; // Kp is pow2 -> shifts
                const float v = (k < K) ? Wg[k * Nn + n] : 0.0f;
                dst[n * Kp + k] = (_Float16)v;
            }
            for (int idx = tid; idx < Nn; idx += 256)
                biasL[LBO[l] + idx] = LB[l][idx];
        }
    }

    // zero the 2x256 output accumulator
    for (int idx = tid; idx < 2 * OUTF; idx += 256) accOut[idx] = 0.0f;

    // ---- Stage 1: triplet features for this WG's 128 rows ---------------
    if (tid < TM) {
        const int r   = blockIdx.x * TM + tid;
        const int b   = r / (NATOM * TRI);
        int rem       = r - b * (NATOM * TRI);
        const int i   = rem / TRI;
        int t         = rem - i * TRI;
        int oa = 0;
        while (t >= (30 - oa)) { t -= (30 - oa); ++oa; }
        const int ob = oa + 1 + t;
        const int j  = oa + (oa >= i ? 1 : 0);
        const int k  = ob + (ob >= i ? 1 : 0);

        const float Rij = D[(b * NATOM + i) * NATOM + j];
        const float Rik = D[(b * NATOM + i) * NATOM + k];
        const float Rjk = D[(b * NATOM + j) * NATOM + k];
        const float zi = Z[b * NATOM + i], zj = Z[b * NATOM + j], zk = Z[b * NATOM + k];

        const float ci = (Rij * Rij + Rik * Rik - Rjk * Rjk) / fmaxf(2.0f * Rij * Rik, 1e-10f);
        const float cj = (Rij * Rij + Rjk * Rjk - Rik * Rik) / fmaxf(2.0f * Rij * Rjk, 1e-10f);
        const float ck = (Rik * Rik + Rjk * Rjk - Rij * Rij) / fmaxf(2.0f * Rik * Rjk, 1e-10f);

        float f[9];
        {
            const float g0 = Rij + Rik + Rjk;
            const float g1 = Rij * Rik + Rij * Rjk + Rik * Rjk;
            const float g2 = Rij * Rik * Rjk;
            const float gn = sqrtf(g0 * g0 + g1 * g1 + g2 * g2) + 1e-7f;
            f[0] = g0 / gn; f[1] = g1 / gn; f[2] = g2 / gn;
            const float c0 = zi + zj + zk;
            const float c1 = ci + cj + ck;
            const float c2 = zi * (zj + zk) + zj * zk - ci * (cj + ck) - cj * ck;
            const float c3 = zi * (cj + ck) + ci * (zj + zk) + zj * ck + cj * zk;
            const float c4 = zi * (zj * zk - cj * ck) - ci * (zj * ck + cj * zk);
            const float c5 = zi * (zj * ck + cj * zk) + ci * (zj * zk - cj * ck);
            const float cn = sqrtf(c0*c0 + c1*c1 + c2*c2 + c3*c3 + c4*c4 + c5*c5) + 1e-7f;
            f[3] = c0 / cn; f[4] = c1 / cn; f[5] = c2 / cn;
            f[6] = c3 / cn; f[7] = c4 / cn; f[8] = c5 / cn;
        }
        const int w = tid >> 4, m = tid & 15;
        _Float16* ain = (_Float16*)(actBase + w * ACT_PER_WAVE);
#pragma unroll
        for (int c = 0; c < 9; ++c) ain[m * 32 + c] = (_Float16)f[c];
#pragma unroll
        for (int c = 9; c < 32; ++c) ain[m * 32 + c] = (_Float16)0.0f;

        const float within = (Rij < CUTOFF && Rik < CUTOFF) ? 1.0f : 0.0f;
        smoothArr[tid] = f_cut(Rij) * f_cut(Rik) * within;
        const int ro = b * NATOM + i;
        rowoutArr[tid] = ro;

        // group id of this tile's first row (computed independently per thread)
        const int r0   = blockIdx.x * TM;
        const int b0i  = r0 / (NATOM * TRI);
        const int i0   = (r0 - b0i * (NATOM * TRI)) / TRI;
        giArr[tid] = (ro != (b0i * NATOM + i0)) ? 1 : 0;
    }

    __syncthreads();

    // ---- Stage 2: per-wave 16-row MLP via WMMA --------------------------
    const int wave = tid >> 5;
    const int lane = tid & 31;
    char* a = actBase + wave * ACT_PER_WAVE;
    _Float16* ain = (_Float16*)(a);
    _Float16* X   = (_Float16*)(a + 1024);
    _Float16* H   = (_Float16*)(a + 3072);
    _Float16* BLK = (_Float16*)(a + 5120);
    _Float16* B5  = (_Float16*)(a + 7168);

    // x_res = tanh(in @ W0 + b0)
    wmma_layer(ain, 32, X,   64, wT + OW0, biasL + 0,   nullptr, lane);
    // blk1 = tanh(x_res @ W1 + b1) + x_res
    wmma_layer(X,   64, BLK, 64, wT + OW1, biasL + 64,  X,       lane);
    // h = tanh(blk1 @ W2 + b2)
    wmma_layer(BLK, 64, H,   64, wT + OW2, biasL + 128, nullptr, lane);
    // h = tanh(h @ W3 + b3)
    wmma_layer(H,   64, X,   64, wT + OW3, biasL + 192, nullptr, lane);
    // blk2 = tanh(h @ W4 + b4) + blk1   (elementwise read-then-write on BLK)
    wmma_layer(X,   64, BLK, 64, wT + OW4, biasL + 256, BLK,     lane);
    // blk3 = tanh(blk2 @ W5 + b5)
    wmma_layer(BLK, 64, B5, 128, wT + OW5, biasL + 320, nullptr, lane);
    // 2-group LDS reduction of tanh(blk3 @ W6 + b6) * smooth
    wmma_final(B5, wT + OW6, biasL + 448, smoothArr + wave * 16,
               giArr + wave * 16, accOut, lane);

    __syncthreads();

    // ---- Stage 3: flush 2x256 partial sums with global atomics ----------
    for (int idx = tid; idx < 2 * OUTF; idx += 256) {
        const int gi = idx >> 8;   // 0 or 1
        const int n  = idx & 255;
        const int ro = rowoutArr[gi ? (TM - 1) : 0];
        // if the tile has a single group, accOut[256..511] is all zeros
        unsafeAtomicAdd(out + ro * OUTF + n, accOut[idx]);
    }
}

__global__ void zero_out_kernel(float* out, int n) {
    const int i = blockIdx.x * blockDim.x + threadIdx.x;
    if (i < n) out[i] = 0.0f;
}

extern "C" void kernel_launch(void* const* d_in, const int* in_sizes, int n_in,
                              void* d_out, int out_size, void* d_ws, size_t ws_size,
                              hipStream_t stream) {
    const float* D  = (const float*)d_in[0];
    const float* Z  = (const float*)d_in[1];
    const float* W0 = (const float*)d_in[2];  const float* b0 = (const float*)d_in[3];
    const float* W1 = (const float*)d_in[4];  const float* b1 = (const float*)d_in[5];
    const float* W2 = (const float*)d_in[6];  const float* b2 = (const float*)d_in[7];
    const float* W3 = (const float*)d_in[8];  const float* b3 = (const float*)d_in[9];
    const float* W4 = (const float*)d_in[10]; const float* b4 = (const float*)d_in[11];
    const float* W5 = (const float*)d_in[12]; const float* b5 = (const float*)d_in[13];
    const float* W6 = (const float*)d_in[14]; const float* b6 = (const float*)d_in[15];
    float* out = (float*)d_out;

    zero_out_kernel<<<(out_size + 255) / 256, 256, 0, stream>>>(out, out_size);

    const int nblocks = ROWS_TOTAL / TM; // 3720
    aev_triplet_kernel<<<nblocks, 256, SMEM_TOTAL, stream>>>(
        D, Z, W0, b0, W1, b1, W2, b2, W3, b3, W4, b4, W5, b5, W6, b6, out);
}
